// SlidingWindowAttention_44968307589534
// MI455X (gfx1250) — matrix operand order repaired and checked by reading, not compiled, not run on hardware
//
#include <hip/hip_runtime.h>
#include <hip/hip_bf16.h>
#include <cstdint>

// ---------------------------------------------------------------------------
// SlidingWindowAttention on gfx1250 (MI455X).
// All matmuls on v_wmma_f32_16x16x32_bf16; contiguous b128 operand fetches;
// GEMM waves register-blocked to 16x64 output strips (4 WMMA accumulators).
// B=1, T=2048, C=512, H=8, KVH=2, D=64, W=128, GROUPS=4, SCALE=1/8
// ---------------------------------------------------------------------------

#define T_LEN 2048
#define CDIM  512
#define NHEAD 8
#define NKV   2
#define HD    64
#define WIN   128
#define QKVN  768            // 512 + 128 + 128
#define PADF  128            // front zero rows (window reach)
#define PADB  32             // back zero rows (last P.V chunk overshoot)
#define ROWS  (PADF + T_LEN + PADB)   // 2208

typedef __attribute__((ext_vector_type(16))) __bf16 v16bf;
typedef __attribute__((ext_vector_type(8)))  __bf16 v8bf;
typedef __attribute__((ext_vector_type(8)))  float  v8f;

static __device__ inline v16bf ld16(const __bf16* p) { return *(const v16bf*)p; }
static __device__ inline v16bf ld8x2(const __bf16* p0, const __bf16* p1) {
    v8bf lo = *(const v8bf*)p0;
    v8bf hi = *(const v8bf*)p1;
    return __builtin_shufflevector(lo, hi, 0,1,2,3,4,5,6,7,8,9,10,11,12,13,14,15);
}

// ---------------------------------------------------------------------------
// Elementwise prep kernels
// ---------------------------------------------------------------------------
__global__ void cvt_bf16_kernel(const float* __restrict__ src, __bf16* __restrict__ dst, int n)
{
    int i = blockIdx.x * 256 + threadIdx.x;
    if (i < n) dst[i] = (__bf16)src[i];
}

// W [K x N] row-major f32 -> Wt [N x K] bf16
__global__ void transpose_cvt_kernel(const float* __restrict__ W, __bf16* __restrict__ Wt,
                                     int K, int N)
{
    int i = blockIdx.x * 256 + threadIdx.x;
    if (i >= K * N) return;
    int k = i / N, n = i % N;
    Wt[(size_t)n * K + k] = (__bf16)W[i];
}

__global__ void zero64_kernel(uint64_t* __restrict__ p, int n64)
{
    int i = blockIdx.x * 256 + threadIdx.x;
    if (i < n64) p[i] = 0ull;
}

// ---------------------------------------------------------------------------
// 16x64-strip bf16 WMMA GEMM: C[M,N] = A[M,K] @ B[K,N]
// A bf16 row-major (lda); Bt bf16 TRANSPOSED [N][K] (ldb = K); C f32 (ldc).
// One wave per 16(M) x 64(N) strip: 4 accumulators reuse each A fetch 4x.
// blockDim = (32, 8); strips = gridDim.x * 8; Nblk = N/64.
// ---------------------------------------------------------------------------
__global__ void gemm16x64_wmma(const __bf16* __restrict__ A, const __bf16* __restrict__ Bt,
                               float* __restrict__ C, int Nblk, int K,
                               int lda, int ldb, int ldc)
{
    const int lane = threadIdx.x;            // 0..31
    const int wave = threadIdx.y;            // 0..7
    const int strip = blockIdx.x * 8 + wave;
    const int mt = strip / Nblk;
    const int nb = strip % Nblk;
    const int half = lane >> 4;              // 0 or 1
    const int ln   = lane & 15;
    const int m0 = mt * 16, n0 = nb * 64;

    const __bf16* Arow = A + (size_t)(m0 + ln) * lda;
    const __bf16* Brow0 = Bt + (size_t)(n0 + ln) * ldb;

    v8f acc[4];
    #pragma unroll
    for (int nt = 0; nt < 4; ++nt) acc[nt] = (v8f){};

    for (int kk = 0; kk < K; kk += 32) {
        // A layout (16-bit, 16x32): v[i]=K[16h+i] (contiguous, matches B)
        v16bf a = ld16(Arow + kk + 16 * half);
        #pragma unroll
        for (int nt = 0; nt < 4; ++nt) {
            // B layout (32x16): lane col = ln; v[i] = K[16h+i] (contiguous in Bt row)
            v16bf b = ld16(Brow0 + (size_t)(nt * 16) * ldb + kk + 16 * half);
            acc[nt] = __builtin_amdgcn_wmma_f32_16x16x32_bf16(false, a, false, b,
                                                              (short)0, acc[nt], false, false);
        }
    }
    #pragma unroll
    for (int nt = 0; nt < 4; ++nt)
        #pragma unroll
        for (int r = 0; r < 8; ++r)
            C[(size_t)(m0 + r + 8 * half) * ldc + n0 + nt * 16 + ln] = acc[nt][r];
}

// ---------------------------------------------------------------------------
// RoPE + bf16 pack. QKV fp32 [T,768] ->
//   Qb [H][T][64], Kb [KVH][ROWS][64] (rows PADF..PADF+T-1), Vt [KVH][64][ROWS].
// ---------------------------------------------------------------------------
__global__ void rope_pack_kernel(const float* __restrict__ QKV,
                                 __bf16* __restrict__ Qo,
                                 __bf16* __restrict__ Ko,
                                 __bf16* __restrict__ Vt)
{
    int idx = blockIdx.x * 256 + threadIdx.x;
    if (idx >= T_LEN * QKVN) return;
    int t = idx / QKVN;
    int c = idx % QKVN;
    float v = QKV[idx];

    if (c >= 640) {                       // V: transpose into [kvh][d][row]
        int kh = (c - 640) / HD, d = (c - 640) % HD;
        Vt[((size_t)(kh * HD + d)) * ROWS + PADF + t] = (__bf16)v;
    } else {
        bool isQ = (c < 512);
        int cc = isQ ? c : (c - 512);
        int hh = cc / HD, d = cc % HD;
        int i = d & 31;                   // emb = concat(freqs, freqs)
        float inv_freq = __expf(-(float)(2 * i) * (9.210340371976184f / 64.0f)); // 10000^(-2i/64)
        float ang = (float)t * inv_freq;
        float sn, cs;
        __sincosf(ang, &sn, &cs);
        int pd = (d < 32) ? (d + 32) : (d - 32);
        float partner = QKV[(size_t)t * QKVN + (isQ ? 0 : 512) + hh * HD + pd];
        float rot = (d < 32) ? -partner : partner;
        float outv = v * cs + rot * sn;
        if (isQ) Qo[((size_t)hh * T_LEN + t) * HD + d] = (__bf16)outv;
        else     Ko[((size_t)hh * ROWS + PADF + t) * HD + d] = (__bf16)outv;
    }
}

// ---------------------------------------------------------------------------
// Sliding-window attention. One wave per (head, 16-query tile).
// Queries [t0, t0+15] attend keys [t0-127, t0+15] => 9 key tiles of 16.
// K/V fetched through zero-padded rows: no predication anywhere.
// blockDim = (32, 8); grid = H*(T/16)/8 = 128 blocks.
// ---------------------------------------------------------------------------
__global__ void attn_wmma_kernel(const __bf16* __restrict__ Qb,
                                 const __bf16* __restrict__ Kb,
                                 const __bf16* __restrict__ Vt,
                                 __bf16* __restrict__ AO)
{
    __shared__ __bf16 Plds[8 * 16 * 176];    // per-wave 16x176 bf16 P staging
    const int lane = threadIdx.x;
    const int wave = threadIdx.y;
    const int tile = blockIdx.x * 8 + wave;  // 0..1023
    const int qt = tile & 127;               // query tile
    const int h  = tile >> 7;                // head
    const int kvh = h >> 2;                  // GROUPS = 4
    const int half = lane >> 4;
    const int ln   = lane & 15;
    const int t0 = qt * 16;
    const int keyfirst = (qt - 8) * 16;      // may be negative; PADF covers it
    __bf16* P = &Plds[wave * 16 * 176];

    const __bf16* Qh = Qb + (size_t)h * T_LEN * HD + (size_t)(t0 + ln) * HD;
    const __bf16* Kh = Kb + (size_t)kvh * ROWS * HD;
    const __bf16* Vh = Vt + (size_t)kvh * HD * ROWS;

    // prefetch V strips (gfx1250 global_prefetch path) while scores run
    #pragma unroll
    for (int nt = 0; nt < 4; ++nt)
        __builtin_prefetch(Vh + (size_t)(nt * 16 + ln) * ROWS + (PADF + keyfirst), 0, 1);

    // ---- Q in A-layout, two K-chunks over D=64 (contiguous vector loads) ----
    v16bf qa[2];
    #pragma unroll
    for (int c = 0; c < 2; ++c)
        qa[c] = ld16(Qh + c * 32 + 16 * half);

    // ---- scores: 9 key tiles, S = Q @ K^T (2 WMMAs each over D) ----
    float sc[9][8];
    #pragma unroll
    for (int kt = 0; kt < 9; ++kt) {
        v8f s = {};
        const __bf16* Krow = Kh + (size_t)(PADF + keyfirst + kt * 16 + ln) * HD;
        #pragma unroll
        for (int c = 0; c < 2; ++c) {
            v16bf b = ld16(Krow + c * 32 + 16 * half);   // d contiguous
            s = __builtin_amdgcn_wmma_f32_16x16x32_bf16(false, qa[c], false, b,
                                                        (short)0, s, false, false);
        }
        #pragma unroll
        for (int r = 0; r < 8; ++r) {
            int t = t0 + r + 8 * half;       // query row for this C slot
            int j = keyfirst + kt * 16 + ln; // key col (j<0 rows are zero => score 0)
            bool inwin = (j <= t) && (j >= t - (WIN - 1));
            sc[kt][r] = inwin ? s[r] * 0.125f : -1e30f;  // SCALE = 1/8
        }
    }

    // ---- softmax (masked), row reduce over the 16 lanes of each half ----
    float rs[8];
    #pragma unroll
    for (int r = 0; r < 8; ++r) {
        float m = sc[0][r];
        #pragma unroll
        for (int kt = 1; kt < 9; ++kt) m = fmaxf(m, sc[kt][r]);
        for (int o = 1; o < 16; o <<= 1) m = fmaxf(m, __shfl_xor(m, o, 32));
        float sum = 0.0f;
        #pragma unroll
        for (int kt = 0; kt < 9; ++kt) {
            float p = __expf(sc[kt][r] - m);
            sc[kt][r] = p;
            sum += p;
        }
        for (int o = 1; o < 16; o <<= 1) sum += __shfl_xor(sum, o, 32);
        rs[r] = sum;
    }

    // ---- stage P (C-layout regs -> row-major LDS); zero cols 144..159 ----
    #pragma unroll
    for (int kt = 0; kt < 9; ++kt)
        #pragma unroll
        for (int r = 0; r < 8; ++r)
            P[(r + 8 * half) * 176 + kt * 16 + ln] = (__bf16)sc[kt][r];
    #pragma unroll
    for (int r = 0; r < 8; ++r)
        P[(r + 8 * half) * 176 + 144 + ln] = (__bf16)0.0f;
    asm volatile("s_wait_dscnt 0x0" ::: "memory");

    // ---- O = P @ V : 5 K-chunks of 32, 4 N-tiles over D (all contiguous) ----
    v8f acc[4];
    #pragma unroll
    for (int nt = 0; nt < 4; ++nt) acc[nt] = (v8f){};
    #pragma unroll
    for (int c = 0; c < 5; ++c) {
        v16bf pa = ld16(P + ln * 176 + c * 32 + 16 * half);
        #pragma unroll
        for (int nt = 0; nt < 4; ++nt) {
            // B: rows = keys (contiguous along Vt row), col d = nt*16 + ln
            const __bf16* Vrow = Vh + (size_t)(nt * 16 + ln) * ROWS
                                    + (PADF + keyfirst) + c * 32 + 16 * half;
            v16bf b = ld16(Vrow);
            acc[nt] = __builtin_amdgcn_wmma_f32_16x16x32_bf16(false, pa, false, b,
                                                              (short)0, acc[nt], false, false);
        }
    }

    // ---- normalize and store AO[t][h*64 + d] bf16 ----
    #pragma unroll
    for (int nt = 0; nt < 4; ++nt)
        #pragma unroll
        for (int r = 0; r < 8; ++r) {
            int t = t0 + r + 8 * half;
            AO[(size_t)t * CDIM + h * HD + nt * 16 + ln] = (__bf16)(acc[nt][r] / rs[r]);
        }
}

// ---------------------------------------------------------------------------
// Launch pipeline
// ---------------------------------------------------------------------------
extern "C" void kernel_launch(void* const* d_in, const int* in_sizes, int n_in,
                              void* d_out, int out_size, void* d_ws, size_t ws_size,
                              hipStream_t stream)
{
    const float* x  = (const float*)d_in[0];
    const float* Wq = (const float*)d_in[1];
    const float* Wk = (const float*)d_in[2];
    const float* Wv = (const float*)d_in[3];
    const float* Wo = (const float*)d_in[4];
    float* out = (float*)d_out;

    // workspace carve-up
    char* ws = (char*)d_ws;
    float*  QKV = (float*)ws;                                   // 2048*768 f32
    __bf16* xb  = (__bf16*)(ws + (size_t)T_LEN * QKVN * 4);     // 2048*512
    __bf16* Wqt = xb  + (size_t)T_LEN * CDIM;                   // 512*512  [N][K]
    __bf16* Wkt = Wqt + (size_t)CDIM * CDIM;                    // 128*512
    __bf16* Wvt = Wkt + (size_t)(NKV * HD) * CDIM;              // 128*512
    __bf16* Wot = Wvt + (size_t)(NKV * HD) * CDIM;              // 512*512
    __bf16* Qb  = Wot + (size_t)CDIM * CDIM;                    // [H][T][64]
    __bf16* Kb  = Qb  + (size_t)NHEAD * T_LEN * HD;             // [KVH][ROWS][64]
    __bf16* Vt  = Kb  + (size_t)NKV * ROWS * HD;                // [KVH][64][ROWS]
    __bf16* AO  = Vt  + (size_t)NKV * HD * ROWS;                // [T][512]

    dim3 blk(32, 8);

    // --- prep: x -> bf16; weights -> bf16 transposed; zero K/V pad buffers ---
    cvt_bf16_kernel<<<(T_LEN * CDIM) / 256, 256, 0, stream>>>(x, xb, T_LEN * CDIM);
    transpose_cvt_kernel<<<(CDIM * CDIM) / 256, 256, 0, stream>>>(Wq, Wqt, CDIM, CDIM);
    transpose_cvt_kernel<<<(CDIM * NKV * HD) / 256, 256, 0, stream>>>(Wk, Wkt, CDIM, NKV * HD);
    transpose_cvt_kernel<<<(CDIM * NKV * HD) / 256, 256, 0, stream>>>(Wv, Wvt, CDIM, NKV * HD);
    transpose_cvt_kernel<<<(CDIM * CDIM) / 256, 256, 0, stream>>>(Wo, Wot, CDIM, CDIM);
    {   // Kb and Vt are adjacent: zero both (pads + interior) in one pass
        int n64 = (int)((size_t)(NKV * ROWS * HD + NKV * HD * ROWS) * 2 / 8);
        zero64_kernel<<<(n64 + 255) / 256, 256, 0, stream>>>((uint64_t*)Kb, n64);
    }

    // --- QKV projections: xb @ {Wq,Wk,Wv} -> QKV f32 [T,768] ---
    // strips: M/16=128 x N/64 = {8,2,2}
    gemm16x64_wmma<<<128, blk, 0, stream>>>(xb, Wqt, QKV + 0,   8, CDIM, CDIM, CDIM, QKVN);
    gemm16x64_wmma<<< 32, blk, 0, stream>>>(xb, Wkt, QKV + 512, 2, CDIM, CDIM, CDIM, QKVN);
    gemm16x64_wmma<<< 32, blk, 0, stream>>>(xb, Wvt, QKV + 640, 2, CDIM, CDIM, CDIM, QKVN);

    // --- RoPE + pack into padded per-head layouts ---
    rope_pack_kernel<<<(T_LEN * QKVN) / 256, 256, 0, stream>>>(QKV, Qb, Kb, Vt);

    // --- sliding-window attention (1024 wave-tiles) ---
    attn_wmma_kernel<<<128, blk, 0, stream>>>(Qb, Kb, Vt, AO);

    // --- output projection: AO @ Wo -> out f32 ---
    gemm16x64_wmma<<<128, blk, 0, stream>>>(AO, Wot, out, 8, CDIM, CDIM, CDIM, CDIM);
}
